// PLE_68659347194005
// MI455X (gfx1250) — matrix operand order repaired and checked by reading, not compiled, not run on hardware
//
#include <hip/hip_runtime.h>

// ---------------- problem constants (match reference) ----------------
#define BB   16384
#define FF   32
#define VV   50000
#define DEMB 16
#define DIN  512     // FF*DEMB
#define EE   6
#define HH1  512
#define HH2  256
#define TT1  128
#define TT2  64

typedef __attribute__((ext_vector_type(16))) _Float16 v16h;
typedef __attribute__((ext_vector_type(8)))  _Float16 v8h;
typedef __attribute__((ext_vector_type(8)))  float    v8f;

// =====================================================================
// Generic grouped GEMM:  out[e] = act( A[e'] @ W[e]^T + bias[e] )
//   A  : f16, [.., M, K] row-major, lda = K
//   Wt : f16, [E][N][K]  (pre-transposed so B-fragment K is contiguous)
//   block = 256 thr = 8 waves; wave -> 32 rows x 64 cols (2x4 WMMA tiles)
//   grid = (M/256, N/64, E)
// a_mode: 0 shared A, 1 A + e*M*lda, 2 PLE group map (0,0,1,1,2,2)
// act:    0 relu, 1 sigmoid
// =====================================================================
__global__ __launch_bounds__(256) void gemm_wmma_f16(
    const _Float16* __restrict__ A, int a_mode, int lda,
    const _Float16* __restrict__ Wt, const float* __restrict__ bias,
    int M, int N, int K,
    float* __restrict__ outF, int ldoF, long long strideF,
    _Float16* __restrict__ outH, int ldoH, long long strideH,
    int act)
{
  const int e    = blockIdx.z;
  const int wave = threadIdx.x >> 5;
  const int lane = threadIdx.x & 31;
  const int m0   = blockIdx.x * 256 + wave * 32;
  const int n0   = blockIdx.y * 64;

  size_t aoff = 0;
  if (a_mode == 1)      aoff = (size_t)e * (size_t)M * (size_t)lda;
  else if (a_mode == 2) { int g = (e < 2) ? 0 : ((e < 4) ? 1 : 2);
                          aoff = (size_t)g * (size_t)M * (size_t)lda; }

  const _Float16* __restrict__ Ae = A  + aoff;
  const _Float16* __restrict__ We = Wt + (size_t)e * (size_t)N * (size_t)K;
  const float*    __restrict__ be = bias + (size_t)e * (size_t)N;

  // ISA 7.12.2 layouts (wave32):
  //  A 16x32 f16: lane&15 = row M, lane>>4 selects K 0..15 vs 16..31 (contig in K; code's order, not ISA's)
  //  B 32x16 f16: lane&15 = col N, lane>>4 selects K 0..15 vs 16..31 (contig in K; code's order, not ISA's)
  //  C 16x16 f32: vgpr r -> row r + 8*(lane>>4), col = lane&15
  const int row   = lane & 15;
  const int khalf = lane >> 4;
  const int col   = lane & 15;
  const int ksec  = lane >> 4;

  v8f acc[2][4];
#pragma unroll
  for (int mt = 0; mt < 2; ++mt)
#pragma unroll
    for (int i = 0; i < 4; ++i)
#pragma unroll
      for (int r = 0; r < 8; ++r) acc[mt][i][r] = 0.0f;

  const _Float16* ap0 = Ae + (size_t)(m0 + row) * lda + khalf * 16;
  const _Float16* ap1 = ap0 + (size_t)16 * (size_t)lda;
  const _Float16* bp  = We + (size_t)(n0 + col) * K + ksec * 16;
  const size_t bstride = (size_t)16 * (size_t)K;   // 16 columns of Wt

  for (int k0 = 0; k0 < K; k0 += 32) {
    __builtin_prefetch(ap0 + k0 + 128, 0, 3);      // global_prefetch_b8 on A streams
    __builtin_prefetch(ap1 + k0 + 128, 0, 3);
    union { v16h v; v8h h[2]; } ua0, ua1;
    ua0.h[0] = *(const v8h*)(ap0 + k0);            // K = khalf*16 + 0..7
    ua0.h[1] = *(const v8h*)(ap0 + k0 + 8);        // K = khalf*16 + 8..15
    ua1.h[0] = *(const v8h*)(ap1 + k0);
    ua1.h[1] = *(const v8h*)(ap1 + k0 + 8);
    const v16h a0 = ua0.v;
    const v16h a1 = ua1.v;
#pragma unroll
    for (int nt = 0; nt < 4; ++nt) {
      const v16h b = *(const v16h*)(bp + (size_t)nt * bstride + k0);
      acc[0][nt] = __builtin_amdgcn_wmma_f32_16x16x32_f16(
          false, a0, false, b, (short)0, acc[0][nt], false, false);
      acc[1][nt] = __builtin_amdgcn_wmma_f32_16x16x32_f16(
          false, a1, false, b, (short)0, acc[1][nt], false, false);
    }
  }

  const int rbase = (lane >> 4) * 8;
#pragma unroll
  for (int mt = 0; mt < 2; ++mt) {
#pragma unroll
    for (int nt = 0; nt < 4; ++nt) {
      const int n = n0 + nt * 16 + col;
      const float bv = be[n];
#pragma unroll
      for (int r = 0; r < 8; ++r) {
        float v = acc[mt][nt][r] + bv;
        if (act == 0)      v = v > 0.0f ? v : 0.0f;
        else               v = 1.0f / (1.0f + __expf(-v));
        const size_t gr = (size_t)(m0 + mt * 16 + rbase + r);
        if (outF) outF[(size_t)e * strideF + gr * (size_t)ldoF + n] = v;
        if (outH) outH[(size_t)e * strideH + gr * (size_t)ldoH + n] = (_Float16)v;
      }
    }
  }
}

// ------------- fp32 -> f16 weight convert + transpose  [E][K][N] -> [E][N][K]
__global__ void conv_w(const float* __restrict__ W, _Float16* __restrict__ Wt,
                       int E, int K, int N)
{
  long long idx = (long long)blockIdx.x * blockDim.x + threadIdx.x;
  long long total = (long long)E * K * N;
  if (idx >= total) return;
  int e   = (int)(idx / ((long long)K * N));
  int rem = (int)(idx % ((long long)K * N));
  int n = rem / K, k = rem % K;
  Wt[idx] = (_Float16)W[((size_t)e * K + k) * N + n];
}

// ------------- embedding gather: emb[f, id, :16] -> X0[b, f*16:..] (f16)
__global__ void embed_gather(const int* __restrict__ ids,
                             const float* __restrict__ emb,
                             _Float16* __restrict__ X0)
{
  int idx = blockIdx.x * blockDim.x + threadIdx.x;   // one thread per (b,f)
  if (idx >= BB * FF) return;
  int b = idx / FF, f = idx % FF;
  int id = ids[idx];
  const float* src = emb + ((size_t)f * VV + (size_t)id) * DEMB;
  _Float16* dst = X0 + (size_t)b * DIN + f * DEMB;
#pragma unroll
  for (int d = 0; d < DEMB; ++d) dst[d] = (_Float16)src[d];
}

// ------------- small helpers -----------------------------------------
template<int NN>
__device__ inline void wave_allreduce(float (&v)[NN]) {
#pragma unroll
  for (int j = 0; j < NN; ++j)
#pragma unroll
    for (int off = 16; off > 0; off >>= 1)
      v[j] += __shfl_xor(v[j], off, 32);
}
template<int NN>
__device__ inline void softmax_small(float (&v)[NN]) {
  float m = v[0];
#pragma unroll
  for (int i = 1; i < NN; ++i) m = fmaxf(m, v[i]);
  float s = 0.0f;
#pragma unroll
  for (int i = 0; i < NN; ++i) { v[i] = __expf(v[i] - m); s += v[i]; }
  float inv = 1.0f / s;
#pragma unroll
  for (int i = 0; i < NN; ++i) v[i] *= inv;
}

// ------------- level-1 gates + expert mixture (one wave per row) ------
__global__ __launch_bounds__(256) void gate_combine1(
    const _Float16* __restrict__ X0,   // [B, 512] f16
    const float* __restrict__ gWs,     // [2,512,4]
    const float* __restrict__ gbs,     // [2,4]
    const float* __restrict__ gWsh,    // [512,6]
    const float* __restrict__ gbsh,    // [6]
    const float* __restrict__ O,       // [6,B,256] f32 expert outputs
    _Float16* __restrict__ xs)         // [3,B,256] f16 mixed outputs
{
  const int wave = threadIdx.x >> 5, lane = threadIdx.x & 31;
  const int row  = blockIdx.x * 8 + wave;
  if (row >= BB) return;

  float l0[4] = {}, l1[4] = {}, ls[6] = {};
  const _Float16* xp = X0 + (size_t)row * DIN;
  for (int k = lane; k < DIN; k += 32) {
    float x = (float)xp[k];
#pragma unroll
    for (int j = 0; j < 4; ++j) l0[j] += x * gWs[(size_t)k * 4 + j];
#pragma unroll
    for (int j = 0; j < 4; ++j) l1[j] += x * gWs[(size_t)DIN * 4 + (size_t)k * 4 + j];
#pragma unroll
    for (int j = 0; j < 6; ++j) ls[j] += x * gWsh[(size_t)k * 6 + j];
  }
  wave_allreduce(l0); wave_allreduce(l1); wave_allreduce(ls);
#pragma unroll
  for (int j = 0; j < 4; ++j) { l0[j] += gbs[j]; l1[j] += gbs[4 + j]; }
#pragma unroll
  for (int j = 0; j < 6; ++j) ls[j] += gbsh[j];
  softmax_small(l0); softmax_small(l1); softmax_small(ls);

  const size_t es = (size_t)BB * HH2;
  const float* Or = O + (size_t)row * HH2;
  for (int c = lane; c < HH2; c += 32) {
    float o0 = Or[0*es+c], o1 = Or[1*es+c], o2 = Or[2*es+c],
          o3 = Or[3*es+c], o4 = Or[4*es+c], o5 = Or[5*es+c];
    float y0 = l0[0]*o0 + l0[1]*o1 + l0[2]*o4 + l0[3]*o5;              // task0: e{0,1,4,5}
    float y1 = l1[0]*o2 + l1[1]*o3 + l1[2]*o4 + l1[3]*o5;              // task1: e{2,3,4,5}
    float y2 = ls[0]*o0 + ls[1]*o1 + ls[2]*o2 + ls[3]*o3 + ls[4]*o4 + ls[5]*o5;
    xs[0*es + (size_t)row * HH2 + c] = (_Float16)y0;
    xs[1*es + (size_t)row * HH2 + c] = (_Float16)y1;
    xs[2*es + (size_t)row * HH2 + c] = (_Float16)y2;
  }
}

// ------------- level-2 gates + mixture (no shared gate) ---------------
__global__ __launch_bounds__(256) void gate_combine2(
    _Float16* __restrict__ xs,         // [3,B,256]; reads slots 0,1 -> writes slots 0,1
    const float* __restrict__ gWs,     // [2,256,4]
    const float* __restrict__ gbs,     // [2,4]
    const float* __restrict__ O)       // [6,B,256] f32
{
  const int wave = threadIdx.x >> 5, lane = threadIdx.x & 31;
  const int row  = blockIdx.x * 8 + wave;
  if (row >= BB) return;

  const size_t es = (size_t)BB * HH2;
  float l0[4] = {}, l1[4] = {};
  const _Float16* x0 = xs + (size_t)row * HH2;
  const _Float16* x1 = xs + es + (size_t)row * HH2;
  for (int k = lane; k < HH2; k += 32) {
    float a = (float)x0[k], b = (float)x1[k];
#pragma unroll
    for (int j = 0; j < 4; ++j) l0[j] += a * gWs[(size_t)k * 4 + j];
#pragma unroll
    for (int j = 0; j < 4; ++j) l1[j] += b * gWs[(size_t)HH2 * 4 + (size_t)k * 4 + j];
  }
  wave_allreduce(l0); wave_allreduce(l1);
#pragma unroll
  for (int j = 0; j < 4; ++j) { l0[j] += gbs[j]; l1[j] += gbs[4 + j]; }
  softmax_small(l0); softmax_small(l1);

  const float* Or = O + (size_t)row * HH2;
  for (int c = lane; c < HH2; c += 32) {
    float o0 = Or[0*es+c], o1 = Or[1*es+c], o2 = Or[2*es+c],
          o3 = Or[3*es+c], o4 = Or[4*es+c], o5 = Or[5*es+c];
    float y0 = l0[0]*o0 + l0[1]*o1 + l0[2]*o4 + l0[3]*o5;
    float y1 = l1[0]*o2 + l1[1]*o3 + l1[2]*o4 + l1[3]*o5;
    xs[0*es + (size_t)row * HH2 + c] = (_Float16)y0;   // in-place ok: same-row, read-before-write
    xs[1*es + (size_t)row * HH2 + c] = (_Float16)y1;
  }
}

// =====================================================================
extern "C" void kernel_launch(void* const* d_in, const int* in_sizes, int n_in,
                              void* d_out, int out_size, void* d_ws, size_t ws_size,
                              hipStream_t stream)
{
  const int*   x_ids  = (const int*)  d_in[0];
  const float* emb    = (const float*)d_in[1];
  const float* e1_W1  = (const float*)d_in[2];
  const float* e1_b1  = (const float*)d_in[3];
  const float* e1_W2  = (const float*)d_in[4];
  const float* e1_b2  = (const float*)d_in[5];
  const float* g1_Ws  = (const float*)d_in[6];
  const float* g1_bs  = (const float*)d_in[7];
  const float* g1_Wsh = (const float*)d_in[8];
  const float* g1_bsh = (const float*)d_in[9];
  const float* e2_W1  = (const float*)d_in[10];
  const float* e2_b1  = (const float*)d_in[11];
  const float* e2_W2  = (const float*)d_in[12];
  const float* e2_b2  = (const float*)d_in[13];
  const float* g2_Ws  = (const float*)d_in[14];
  const float* g2_bs  = (const float*)d_in[15];
  const float* tw_W1  = (const float*)d_in[16];
  const float* tw_b1  = (const float*)d_in[17];
  const float* tw_W2  = (const float*)d_in[18];
  const float* tw_b2  = (const float*)d_in[19];
  float* out = (float*)d_out;

  // ---- workspace carve-up (256B aligned) ----
  char* ws = (char*)d_ws;
  size_t off = 0;
  auto take = [&](size_t bytes) -> char* {
    char* p = ws + off;
    off = (off + bytes + 255) & ~(size_t)255;
    return p;
  };
  _Float16* X0    = (_Float16*)take((size_t)BB * DIN * 2);
  _Float16* e1W1t = (_Float16*)take((size_t)EE * HH1 * DIN * 2);
  _Float16* e1W2t = (_Float16*)take((size_t)EE * HH2 * HH1 * 2);
  _Float16* e2W1t = (_Float16*)take((size_t)EE * HH1 * HH2 * 2);
  _Float16* e2W2t = (_Float16*)take((size_t)EE * HH2 * HH1 * 2);
  _Float16* twW1t = (_Float16*)take((size_t)2 * TT1 * HH2 * 2);
  _Float16* twW2t = (_Float16*)take((size_t)2 * TT2 * TT1 * 2);
  _Float16* Hbf   = (_Float16*)take((size_t)EE * BB * HH1 * 2);   // reused: L1 & L2 hidden
  float*    OutF  = (float*)   take((size_t)EE * BB * HH2 * 4);   // reused: L1 & L2 expert out
  _Float16* xsbf  = (_Float16*)take((size_t)3 * BB * HH2 * 2);    // mixed xs (f16)
  _Float16* Thbf  = (_Float16*)take((size_t)2 * BB * TT1 * 2);    // tower hidden

  auto cdiv = [](long long a, long long b) { return (unsigned)((a + b - 1) / b); };

  // ---- weight convert + transpose to [E][N][K] f16 ----
  conv_w<<<cdiv((long long)EE*DIN*HH1, 256), 256, 0, stream>>>(e1_W1, e1W1t, EE, DIN, HH1);
  conv_w<<<cdiv((long long)EE*HH1*HH2, 256), 256, 0, stream>>>(e1_W2, e1W2t, EE, HH1, HH2);
  conv_w<<<cdiv((long long)EE*HH2*HH1, 256), 256, 0, stream>>>(e2_W1, e2W1t, EE, HH2, HH1);
  conv_w<<<cdiv((long long)EE*HH1*HH2, 256), 256, 0, stream>>>(e2_W2, e2W2t, EE, HH1, HH2);
  conv_w<<<cdiv((long long)2*HH2*TT1,  256), 256, 0, stream>>>(tw_W1, twW1t, 2, HH2, TT1);
  conv_w<<<cdiv((long long)2*TT1*TT2,  256), 256, 0, stream>>>(tw_W2, twW2t, 2, TT1, TT2);

  // ---- embedding gather -> X0 f16 ----
  embed_gather<<<cdiv((long long)BB*FF, 256), 256, 0, stream>>>(x_ids, emb, X0);

  const long long sH1 = (long long)BB * HH1;   // [e] stride of Hbf
  const long long sO  = (long long)BB * HH2;   // [e] stride of OutF

  // ---- CGC level 1: h = relu(X0 @ W1 + b1) ; out = relu(h @ W2 + b2) ----
  gemm_wmma_f16<<<dim3(BB/256, HH1/64, EE), 256, 0, stream>>>(
      X0, /*a_mode=*/0, DIN, e1W1t, e1_b1, BB, HH1, DIN,
      (float*)nullptr, 0, 0, Hbf, HH1, sH1, /*act=*/0);
  gemm_wmma_f16<<<dim3(BB/256, HH2/64, EE), 256, 0, stream>>>(
      Hbf, /*a_mode=*/1, HH1, e1W2t, e1_b2, BB, HH2, HH1,
      OutF, HH2, sO, (_Float16*)nullptr, 0, 0, /*act=*/0);

  // ---- level-1 gates + mixture -> xs[0..2] ----
  gate_combine1<<<BB/8, 256, 0, stream>>>(X0, g1_Ws, g1_bs, g1_Wsh, g1_bsh, OutF, xsbf);

  // ---- CGC level 2 (experts read xs via PLE group map) ----
  gemm_wmma_f16<<<dim3(BB/256, HH1/64, EE), 256, 0, stream>>>(
      xsbf, /*a_mode=*/2, HH2, e2W1t, e2_b1, BB, HH1, HH2,
      (float*)nullptr, 0, 0, Hbf, HH1, sH1, /*act=*/0);
  gemm_wmma_f16<<<dim3(BB/256, HH2/64, EE), 256, 0, stream>>>(
      Hbf, /*a_mode=*/1, HH1, e2W2t, e2_b2, BB, HH2, HH1,
      OutF, HH2, sO, (_Float16*)nullptr, 0, 0, /*act=*/0);

  // ---- level-2 gates + mixture -> xs[0..1] (in place) ----
  gate_combine2<<<BB/8, 256, 0, stream>>>(xsbf, g2_Ws, g2_bs, OutF);

  // ---- towers: relu(x @ W1 + b1) ; sigmoid(h @ W2 + b2) -> d_out[B,128] ----
  gemm_wmma_f16<<<dim3(BB/256, TT1/64, 2), 256, 0, stream>>>(
      xsbf, /*a_mode=*/1, HH2, twW1t, tw_b1, BB, TT1, HH2,
      (float*)nullptr, 0, 0, Thbf, TT1, (long long)BB*TT1, /*act=*/0);
  gemm_wmma_f16<<<dim3(BB/256, TT2/64, 2), 256, 0, stream>>>(
      Thbf, /*a_mode=*/1, TT1, twW2t, tw_b2, BB, TT2, TT1,
      out, /*ldoF=*/2*TT2, /*strideF (task col offset)=*/TT2,
      (_Float16*)nullptr, 0, 0, /*act=*/1);
}
